// MTA_sharemoe_58634893525397
// MI455X (gfx1250) — matrix operand order repaired and checked by reading, not compiled, not run on hardware
//
#include <hip/hip_runtime.h>

// ---------- types ----------
typedef __attribute__((ext_vector_type(16))) __bf16   v16bf;
typedef __attribute__((ext_vector_type(8)))  float    v8f;
typedef __attribute__((ext_vector_type(8)))  unsigned short su8;  // 16B
typedef __attribute__((ext_vector_type(4)))  unsigned short su4;  // 8B
typedef int i4v __attribute__((vector_size(16)));                 // b128 payload type

#define BM 128
#define BN 128
#define BK 32
#define LDT 40   // padded LDS K-stride (elements): 80B rows, conflict-free frag reads

// ---------- CDNA5 async global->LDS path (guarded; falls back to reg copies) ----------
#if defined(__has_builtin)
#if __has_builtin(__builtin_amdgcn_global_load_async_to_lds_b128) && \
    __has_builtin(__builtin_amdgcn_s_wait_asynccnt)
#define USE_ASYNC_LDS 1
#endif
#endif

__device__ __forceinline__ void cp_b128(unsigned short* ldst, const unsigned short* gsrc) {
#ifdef USE_ASYNC_LDS
  __builtin_amdgcn_global_load_async_to_lds_b128(
      (__attribute__((address_space(1))) i4v*)(void*)gsrc,
      (__attribute__((address_space(3))) i4v*)(void*)ldst, 0, 0);
#else
  *(su8*)ldst = *(const su8*)gsrc;
#endif
}

__device__ __forceinline__ void stage_wait() {
#ifdef USE_ASYNC_LDS
  __builtin_amdgcn_s_wait_asynccnt(0);
#endif
}

// ---------- helpers ----------
__device__ __forceinline__ unsigned short f32_to_bf16(float f) {
  unsigned int u = __float_as_uint(f);
  unsigned int r = u + 0x7FFFu + ((u >> 16) & 1u);   // round-to-nearest-even
  return (unsigned short)(r >> 16);
}

// branchless tanh-gelu: 0.5x(1+tanh(z)) = x * t/(t+1), t = exp(2z)
__device__ __forceinline__ float gelu_fast(float x) {
  float z2 = 1.5957691216057308f * (x + 0.044715f * x * x * x);
  z2 = fminf(z2, 80.f);
  float t = __expf(z2);
  return x * __fdividef(t, t + 1.f);
}

// A-frag (16x32 bf16, MxK): lane<16 -> K[0..7]+K[16..23]; lane>=16 -> K[8..15]+K[24..31]
__device__ __forceinline__ v16bf afrag(const unsigned short* tile, int r0, int lane) {
  int m   = r0 + (lane & 15);
  int off = ((lane >> 4) & 1) * 8;
  const unsigned short* p = tile + m * LDT + off;
  union { su8 h[2]; v16bf v; } u;
  u.h[0] = *(const su8*)(p);
  u.h[1] = *(const su8*)(p + 16);
  return u.v;
}

// B-frag (32x16 bf16, KxN), tile stored [N][K]: lane<16 -> K[0..7]+K[16..23]; lane>=16 -> K[8..15]+K[24..31]
__device__ __forceinline__ v16bf bfrag(const unsigned short* tile, int n0, int lane) {
  int n  = n0 + (lane & 15);
  int kh = ((lane >> 4) & 1) * 8;
  const unsigned short* p = tile + n * LDT + kh;
  union { su8 h[2]; v16bf v; } u;
  u.h[0] = *(const su8*)(p);
  u.h[1] = *(const su8*)(p + 16);   // K[kh+16..kh+23]
  return u.v;
}

// copy one 128x32 bf16 tile (tile origin g, row stride ldg) into LDS [128][LDT]
__device__ __forceinline__ void tile_to_lds(unsigned short* lds, const unsigned short* g,
                                            size_t ldg, int tid) {
  int r = tid >> 1;
  int c = (tid & 1) * 16;
  const unsigned short* gp = g + (size_t)r * ldg + c;
  unsigned short* lp = lds + r * LDT + c;
  cp_b128(lp, gp);
  cp_b128(lp + 8, gp + 8);
}

// ---------- fp32 -> bf16 (no transpose; for X) ----------
__global__ __launch_bounds__(256) void k_cvt(const float* __restrict__ src,
                                             unsigned short* __restrict__ dst, long n) {
  long i = ((long)blockIdx.x * 256 + threadIdx.x) * 4;
  if (i >= n) return;
  float4 f = *(const float4*)(src + i);
  su4 o;
  o[0] = f32_to_bf16(f.x); o[1] = f32_to_bf16(f.y);
  o[2] = f32_to_bf16(f.z); o[3] = f32_to_bf16(f.w);
  *(su4*)(dst + i) = o;
}

// ---------- fp32 [R][C] -> bf16 [C][R] (one-time weight transpose) ----------
__global__ __launch_bounds__(256) void k_cvtT(const float* __restrict__ in,
                                              unsigned short* __restrict__ out,
                                              int R, int C) {
  __shared__ __align__(16) unsigned short lt[64 * 40];
  const size_t mat = (size_t)blockIdx.z * (size_t)R * (size_t)C;
  in  += mat;
  out += mat;
  const int c0 = blockIdx.x * 64;
  const int r0 = blockIdx.y * 32;
  const int t  = threadIdx.x;
  { // read 32r x 64c coalesced, scatter transposed into LDS [c][r]
    int r  = t >> 3;        // 0..31
    int cc = (t & 7) * 8;   // 0..56
    const float4* src = (const float4*)(in + (size_t)(r0 + r) * C + c0 + cc);
    float4 f0 = src[0], f1 = src[1];
    float v[8] = {f0.x, f0.y, f0.z, f0.w, f1.x, f1.y, f1.z, f1.w};
#pragma unroll
    for (int e = 0; e < 8; ++e) lt[(cc + e) * 40 + r] = f32_to_bf16(v[e]);
  }
  __syncthreads();
  { // write 64c x 32r with 16B vector stores
    int c  = t >> 2;        // 0..63
    int rc = (t & 3) * 8;   // 0..24
    *(su8*)(out + (size_t)(c0 + c) * R + r0 + rc) = *(const su8*)(lt + c * 40 + rc);
  }
}

// ---------- GEMM 1: Act = bf16( gelu(X@W0) * (X@W1) ) ----------
// X:[B*S,D] bf16; W0t/W1t:[(E,)F,D] bf16 (pre-transposed); Act:[B*S,F] bf16
__global__ __launch_bounds__(256) void k_gatedup(
    const unsigned short* __restrict__ X,
    const unsigned short* __restrict__ W0t,
    const unsigned short* __restrict__ W1t,
    const int* __restrict__ labels,
    unsigned short* __restrict__ Act,
    int S, int D, int F) {
  __shared__ __align__(16) unsigned short lA [2][BM * LDT];
  __shared__ __align__(16) unsigned short lB0[2][BN * LDT];
  __shared__ __align__(16) unsigned short lB1[2][BN * LDT];

  const int b    = blockIdx.z;
  const int m0   = blockIdx.x * BM;
  const int n0   = blockIdx.y * BN;
  const int tid  = threadIdx.x;
  const int lane = tid & 31;
  const int wave = tid >> 5;
  const int wm   = wave >> 2;   // 0..1 (64 rows)
  const int wn   = wave & 3;    // 0..3 (32 cols)

  const size_t rowBase = (size_t)b * S + m0;
  size_t wOff = 0;
  if (labels) wOff = (size_t)labels[b] * (size_t)D * (size_t)F;
  const unsigned short* Ag  = X   + rowBase * (size_t)D;
  const unsigned short* B0g = W0t + wOff + (size_t)n0 * D;
  const unsigned short* B1g = W1t + wOff + (size_t)n0 * D;

  const v8f vz = {0.f, 0.f, 0.f, 0.f, 0.f, 0.f, 0.f, 0.f};
  v8f acc0[4][2], acc1[4][2];
#pragma unroll
  for (int i = 0; i < 4; ++i)
#pragma unroll
    for (int j = 0; j < 2; ++j) { acc0[i][j] = vz; acc1[i][j] = vz; }

  const int nst = D / BK;   // 32
  tile_to_lds(lA[0],  Ag,  D, tid);
  tile_to_lds(lB0[0], B0g, D, tid);
  tile_to_lds(lB1[0], B1g, D, tid);

  int cur = 0;
  for (int kt = 0; kt < nst; ++kt) {
    stage_wait();
    __syncthreads();
    if (kt + 1 < nst) {
      int k0 = (kt + 1) * BK;
      tile_to_lds(lA[cur ^ 1],  Ag  + k0, D, tid);
      tile_to_lds(lB0[cur ^ 1], B0g + k0, D, tid);
      tile_to_lds(lB1[cur ^ 1], B1g + k0, D, tid);
    }
    v16bf af[4];
#pragma unroll
    for (int i = 0; i < 4; ++i) af[i] = afrag(lA[cur], wm * 64 + i * 16, lane);
#pragma unroll
    for (int j = 0; j < 2; ++j) {
      v16bf b0 = bfrag(lB0[cur], wn * 32 + j * 16, lane);
      v16bf b1 = bfrag(lB1[cur], wn * 32 + j * 16, lane);
#pragma unroll
      for (int i = 0; i < 4; ++i) {
        acc0[i][j] = __builtin_amdgcn_wmma_f32_16x16x32_bf16(
            false, af[i], false, b0, (short)0, acc0[i][j], false, false);
        acc1[i][j] = __builtin_amdgcn_wmma_f32_16x16x32_bf16(
            false, af[i], false, b1, (short)0, acc1[i][j], false, false);
      }
    }
    cur ^= 1;
  }

  const int nlo = lane & 15;
  const int mhi = (lane >> 4) * 8;
#pragma unroll
  for (int i = 0; i < 4; ++i)
#pragma unroll
    for (int j = 0; j < 2; ++j)
#pragma unroll
      for (int v = 0; v < 8; ++v) {
        int row = m0 + wm * 64 + i * 16 + mhi + v;
        int col = n0 + wn * 32 + j * 16 + nlo;
        float g = gelu_fast(acc0[i][j][v]);
        float l = acc1[i][j][v];
        Act[((size_t)b * S + row) * (size_t)F + col] = f32_to_bf16(g * l);
      }
}

// ---------- GEMM 2: Out = Act @ Wo ----------
// A:[B*S,F] bf16; Wt:[(E,)D,F] bf16 (pre-transposed); Out:[B*S,D] f32
__global__ __launch_bounds__(256) void k_down(
    const unsigned short* __restrict__ A,
    const unsigned short* __restrict__ Wt,
    const int* __restrict__ labels,
    float* __restrict__ Out,
    int S, int F, int D) {
  __shared__ __align__(16) unsigned short lA[2][BM * LDT];
  __shared__ __align__(16) unsigned short lB[2][BN * LDT];

  const int b    = blockIdx.z;
  const int m0   = blockIdx.x * BM;
  const int n0   = blockIdx.y * BN;
  const int tid  = threadIdx.x;
  const int lane = tid & 31;
  const int wave = tid >> 5;
  const int wm   = wave >> 2;
  const int wn   = wave & 3;

  const size_t rowBase = (size_t)b * S + m0;
  size_t wOff = 0;
  if (labels) wOff = (size_t)labels[b] * (size_t)F * (size_t)D;
  const unsigned short* Ag = A  + rowBase * (size_t)F;
  const unsigned short* Bg = Wt + wOff + (size_t)n0 * F;

  const v8f vz = {0.f, 0.f, 0.f, 0.f, 0.f, 0.f, 0.f, 0.f};
  v8f acc[4][2];
#pragma unroll
  for (int i = 0; i < 4; ++i)
#pragma unroll
    for (int j = 0; j < 2; ++j) acc[i][j] = vz;

  const int nst = F / BK;   // 128
  tile_to_lds(lA[0], Ag, F, tid);
  tile_to_lds(lB[0], Bg, F, tid);

  int cur = 0;
  for (int kt = 0; kt < nst; ++kt) {
    stage_wait();
    __syncthreads();
    if (kt + 1 < nst) {
      int k0 = (kt + 1) * BK;
      tile_to_lds(lA[cur ^ 1], Ag + k0, F, tid);
      tile_to_lds(lB[cur ^ 1], Bg + k0, F, tid);
    }
    v16bf af[4];
#pragma unroll
    for (int i = 0; i < 4; ++i) af[i] = afrag(lA[cur], wm * 64 + i * 16, lane);
#pragma unroll
    for (int j = 0; j < 2; ++j) {
      v16bf bf = bfrag(lB[cur], wn * 32 + j * 16, lane);
#pragma unroll
      for (int i = 0; i < 4; ++i)
        acc[i][j] = __builtin_amdgcn_wmma_f32_16x16x32_bf16(
            false, af[i], false, bf, (short)0, acc[i][j], false, false);
    }
    cur ^= 1;
  }

  const int nlo = lane & 15;
  const int mhi = (lane >> 4) * 8;
#pragma unroll
  for (int i = 0; i < 4; ++i)
#pragma unroll
    for (int j = 0; j < 2; ++j)
#pragma unroll
      for (int v = 0; v < 8; ++v) {
        int row = m0 + wm * 64 + i * 16 + mhi + v;
        int col = n0 + wn * 32 + j * 16 + nlo;
        Out[((size_t)b * S + row) * (size_t)D + col] = acc[i][j][v];
      }
}

// ---------- gate: softmax(relu(cat(e0,s0)@g0)@g1) ----------
__global__ __launch_bounds__(256) void k_gate(
    const float* __restrict__ Eout, const float* __restrict__ Sout,
    const float* __restrict__ g0, const float* __restrict__ g1,
    float* __restrict__ gw, int S, int D) {
  __shared__ float xin[2048];
  __shared__ float red[512];
  const int b = blockIdx.x, t = threadIdx.x;
  const float* e0 = Eout + (size_t)b * S * D;  // token 0
  const float* s0 = Sout + (size_t)b * S * D;
  for (int i = t; i < D; i += 256) { xin[i] = e0[i]; xin[D + i] = s0[i]; }
  __syncthreads();

  float h[4] = {0.f, 0.f, 0.f, 0.f};
  for (int i = 0; i < 2 * D; ++i) {
    float xi = xin[i];
    const float* grow = g0 + (size_t)i * D;
#pragma unroll
    for (int q = 0; q < 4; ++q) h[q] += xi * grow[t + q * 256];
  }
  float o0 = 0.f, o1 = 0.f;
#pragma unroll
  for (int q = 0; q < 4; ++q) {
    float hv = h[q] > 0.f ? h[q] : 0.f;
    int j = t + q * 256;
    o0 += hv * g1[j * 2 + 0];
    o1 += hv * g1[j * 2 + 1];
  }
  red[t] = o0; red[256 + t] = o1;
  __syncthreads();
  for (int s2 = 128; s2 > 0; s2 >>= 1) {
    if (t < s2) { red[t] += red[t + s2]; red[256 + t] += red[256 + t + s2]; }
    __syncthreads();
  }
  if (t == 0) {
    float a = red[0], c = red[256];
    float m = fmaxf(a, c);
    float ea = __expf(a - m), ec = __expf(c - m);
    float inv = 1.f / (ea + ec);
    gw[b * 2 + 0] = ea * inv;
    gw[b * 2 + 1] = ec * inv;
  }
}

// ---------- final mix ----------
__global__ __launch_bounds__(256) void k_mix(
    const float* __restrict__ E, const float* __restrict__ Sh,
    const float* __restrict__ gw, float* __restrict__ out, long SD, long n) {
  long i = ((long)blockIdx.x * 256 + threadIdx.x) * 4;
  if (i >= n) return;
  int b = (int)(i / SD);
  float w0 = gw[b * 2 + 0], w1 = gw[b * 2 + 1];
  float4 e = *(const float4*)(E + i);
  float4 s = *(const float4*)(Sh + i);
  float4 r;
  r.x = e.x * w0 + s.x * w1;
  r.y = e.y * w0 + s.y * w1;
  r.z = e.z * w0 + s.z * w1;
  r.w = e.w * w0 + s.w * w1;
  *(float4*)(out + i) = r;
}

// ---------- host ----------
extern "C" void kernel_launch(void* const* d_in, const int* in_sizes, int n_in,
                              void* d_out, int out_size, void* d_ws, size_t ws_size,
                              hipStream_t stream) {
  const float* X    = (const float*)d_in[0];
  const int*   lab  = (const int*)  d_in[1];
  const float* Wi0  = (const float*)d_in[2];
  const float* Wi1  = (const float*)d_in[3];
  const float* Wo   = (const float*)d_in[4];
  const float* sWi0 = (const float*)d_in[5];
  const float* sWi1 = (const float*)d_in[6];
  const float* sWo  = (const float*)d_in[7];
  const float* g0   = (const float*)d_in[8];
  const float* g1   = (const float*)d_in[9];
  float* out = (float*)d_out;

  const int B = 8, S = 2048, D = 1024, F = 4096, E = 5;
  const size_t nX   = (size_t)B * S * D;
  const size_t nW   = (size_t)E * D * F;
  const size_t nsW  = (size_t)D * F;
  const size_t nAct = (size_t)B * S * F;

  char* ws = (char*)d_ws;
  size_t off = 0;
  auto alloc = [&](size_t bytes) -> char* {
    char* p = ws + off;
    off += (bytes + 255) & ~(size_t)255;
    return p;
  };
  unsigned short* Xb    = (unsigned short*)alloc(nX * 2);
  unsigned short* Wi0t  = (unsigned short*)alloc(nW * 2);   // [E][F][D]
  unsigned short* Wi1t  = (unsigned short*)alloc(nW * 2);   // [E][F][D]
  unsigned short* Wot   = (unsigned short*)alloc(nW * 2);   // [E][D][F]
  unsigned short* sWi0t = (unsigned short*)alloc(nsW * 2);  // [F][D]
  unsigned short* sWi1t = (unsigned short*)alloc(nsW * 2);  // [F][D]
  unsigned short* sWot  = (unsigned short*)alloc(nsW * 2);  // [D][F]
  unsigned short* Act   = (unsigned short*)alloc(nAct * 2);
  float*          Eout  = (float*)alloc(nX * 4);
  float*          Sout  = (float*)alloc(nX * 4);
  float*          gw    = (float*)alloc(64);
  (void)ws_size; (void)in_sizes; (void)n_in; (void)out_size;

  // X: plain convert
  k_cvt<<<dim3((unsigned)(nX / 4 / 256)), dim3(256), 0, stream>>>(X, Xb, (long)nX);
  // weights: convert + transpose ([R][C] f32 -> [C][R] bf16)
  k_cvtT<<<dim3(F / 64, D / 32, E), dim3(256), 0, stream>>>(Wi0, Wi0t, D, F);
  k_cvtT<<<dim3(F / 64, D / 32, E), dim3(256), 0, stream>>>(Wi1, Wi1t, D, F);
  k_cvtT<<<dim3(D / 64, F / 32, E), dim3(256), 0, stream>>>(Wo,  Wot,  F, D);
  k_cvtT<<<dim3(F / 64, D / 32, 1), dim3(256), 0, stream>>>(sWi0, sWi0t, D, F);
  k_cvtT<<<dim3(F / 64, D / 32, 1), dim3(256), 0, stream>>>(sWi1, sWi1t, D, F);
  k_cvtT<<<dim3(D / 64, F / 32, 1), dim3(256), 0, stream>>>(sWo,  sWot,  F, D);

  // expert path
  k_gatedup<<<dim3(S / BM, F / BN, B), dim3(256), 0, stream>>>(
      Xb, Wi0t, Wi1t, lab, Act, S, D, F);
  k_down<<<dim3(S / BM, D / BN, B), dim3(256), 0, stream>>>(
      Act, Wot, lab, Eout, S, F, D);

  // shared path (reuses Act)
  k_gatedup<<<dim3(S / BM, F / BN, B), dim3(256), 0, stream>>>(
      Xb, sWi0t, sWi1t, nullptr, Act, S, D, F);
  k_down<<<dim3(S / BM, D / BN, B), dim3(256), 0, stream>>>(
      Act, sWot, nullptr, Sout, S, F, D);

  // gate + mix
  k_gate<<<dim3(B), dim3(256), 0, stream>>>(Eout, Sout, g0, g1, gw, S, D);
  k_mix<<<dim3((unsigned)(nX / 4 / 256)), dim3(256), 0, stream>>>(
      Eout, Sout, gw, out, (long)S * D, (long)nX);
}
